// PAMS_RDB_Conv_in_48850958024803
// MI455X (gfx1250) — 8-bit matrix operand order repaired and checked by reading, compiled, not run on hardware
//
#include <hip/hip_runtime.h>

typedef __attribute__((ext_vector_type(8))) int v8i;

#define NB    16
#define CIN   64
#define GOUT  64
#define HD    128
#define WD    128
#define HP    130
#define WP    130
#define QMAXF 127.0f

#define XQ_IMG_BYTES ((size_t)HP * WP * CIN)
#define XQ_TOTAL     ((size_t)NB * XQ_IMG_BYTES)
#define WQ_BYTES     (9 * 4 * 1024)

__global__ __launch_bounds__(256) void wmax_kernel(const float* __restrict__ w,
                                                   float* __restrict__ wmax) {
    __shared__ float red[256];
    float m = 0.0f;
    for (int i = threadIdx.x; i < GOUT * CIN * 9; i += 256)
        m = fmaxf(m, fabsf(w[i]));
    red[threadIdx.x] = m;
    __syncthreads();
    for (int s = 128; s > 0; s >>= 1) {
        if ((int)threadIdx.x < s)
            red[threadIdx.x] = fmaxf(red[threadIdx.x], red[threadIdx.x + s]);
        __syncthreads();
    }
    if (threadIdx.x == 0) wmax[0] = red[0];
}

__global__ __launch_bounds__(256) void packw_kernel(const float* __restrict__ w,
                                                    const float* __restrict__ wmax,
                                                    signed char* __restrict__ wqp) {
    int idx = blockIdx.x * 256 + threadIdx.x;
    if (idx >= WQ_BYTES) return;
    int tile = idx >> 10;
    int rem  = idx & 1023;
    int lane = rem >> 5;
    int j    = rem & 31;
    int tap  = tile >> 2;
    int mt   = tile & 3;
    int m    = lane & 15;
    int v    = j >> 2, bb = j & 3;
    int k    = ((j >> 4) << 5) + (j & 15) + ((lane >= 16) ? 16 : 0);
    int g    = mt * 16 + m;
    int kh   = tap / 3, kw = tap % 3;
    float wv = w[((g * CIN + k) * 3 + kh) * 3 + kw];
    float s  = QMAXF / wmax[0];
    wqp[idx] = (signed char)(int)rintf(wv * s);
}

__global__ __launch_bounds__(256) void zero_kernel(int4* __restrict__ p, size_t n16) {
    size_t i      = (size_t)blockIdx.x * blockDim.x + threadIdx.x;
    size_t stride = (size_t)gridDim.x * blockDim.x;
    int4 z        = make_int4(0, 0, 0, 0);
    for (; i < n16; i += stride) p[i] = z;
}

__global__ __launch_bounds__(256) void quantx_kernel(const float* __restrict__ x,
                                                     const float* __restrict__ alpha,
                                                     signed char* __restrict__ xq,
                                                     float* __restrict__ out) {
    int idx = blockIdx.x * 256 + threadIdx.x;
    int b   = idx >> 14;
    int hw  = idx & 16383;
    int h   = hw >> 7, w = hw & 127;
    float a   = alpha[0];
    float inv = QMAXF / a;
    float deq = a / QMAXF;

    int packed[16];
#pragma unroll
    for (int c4 = 0; c4 < 16; ++c4) {
        int pk = 0;
#pragma unroll
        for (int jj = 0; jj < 4; ++jj) {
            int c   = c4 * 4 + jj;
            float v = x[(((size_t)b * CIN + c) * HD + h) * WD + w];
            v       = fminf(fmaxf(v, -a), a);
            int q   = (int)rintf(v * inv);
            out[(((size_t)b * 128 + c) * HD + h) * WD + w] = (float)q * deq;
            pk |= (q & 0xff) << (jj * 8);
        }
        packed[c4] = pk;
    }
    int4* dst = (int4*)(xq + (size_t)b * XQ_IMG_BYTES +
                        ((size_t)(h + 1) * WP + (w + 1)) * CIN);
#pragma unroll
    for (int t = 0; t < 4; ++t)
        dst[t] = make_int4(packed[4 * t], packed[4 * t + 1],
                           packed[4 * t + 2], packed[4 * t + 3]);
}

__global__ __launch_bounds__(256) void conv_wmma_kernel(
    const signed char* __restrict__ xq, const signed char* __restrict__ wqp,
    const float* __restrict__ bias, const float* __restrict__ alpha,
    const float* __restrict__ wmax, float* __restrict__ out) {
    const int lane = threadIdx.x & 31;
    const int tile = blockIdx.x * 8 + (threadIdx.x >> 5);
    const int b    = tile >> 9;
    const int rr   = tile & 511;
    const int h    = rr >> 2;
    const int w0   = (rr & 3) << 5;

    const int n  = lane & 15;
    const int hi = lane >> 4;

    const signed char* xbase = xq + (size_t)b * XQ_IMG_BYTES;

    v8i acc0 = {}, acc1 = {}, acc2 = {}, acc3 = {};
    v8i acc4 = {}, acc5 = {}, acc6 = {}, acc7 = {};

#pragma unroll
    for (int kh = 0; kh < 3; ++kh) {
#pragma unroll
        for (int kw = 0; kw < 3; ++kw) {
            const signed char* pB0 =
                xbase + ((size_t)(h + kh) * WP + (w0 + n + kw)) * CIN + hi * 16;
            const signed char* pB1 = pB0 + 16 * CIN;
            const int4 b0lo = *(const int4*)pB0;
            const int4 b0hi = *(const int4*)(pB0 + 32);
            const int4 b1lo = *(const int4*)pB1;
            const int4 b1hi = *(const int4*)(pB1 + 32);
            v8i B0, B1;
            B0[0] = b0lo.x; B0[1] = b0lo.y; B0[2] = b0lo.z; B0[3] = b0lo.w;
            B0[4] = b0hi.x; B0[5] = b0hi.y; B0[6] = b0hi.z; B0[7] = b0hi.w;
            B1[0] = b1lo.x; B1[1] = b1lo.y; B1[2] = b1lo.z; B1[3] = b1lo.w;
            B1[4] = b1hi.x; B1[5] = b1hi.y; B1[6] = b1hi.z; B1[7] = b1hi.w;

            const signed char* pA = wqp + (size_t)((kh * 3 + kw) * 4) * 1024 + lane * 32;
            v8i A0 = *(const v8i*)(pA);
            v8i A1 = *(const v8i*)(pA + 1024);
            v8i A2 = *(const v8i*)(pA + 2048);
            v8i A3 = *(const v8i*)(pA + 3072);

            acc0 = __builtin_amdgcn_wmma_i32_16x16x64_iu8(true, A0, true, B0, acc0, false, false);
            acc4 = __builtin_amdgcn_wmma_i32_16x16x64_iu8(true, A0, true, B1, acc4, false, false);
            acc1 = __builtin_amdgcn_wmma_i32_16x16x64_iu8(true, A1, true, B0, acc1, false, false);
            acc5 = __builtin_amdgcn_wmma_i32_16x16x64_iu8(true, A1, true, B1, acc5, false, false);
            acc2 = __builtin_amdgcn_wmma_i32_16x16x64_iu8(true, A2, true, B0, acc2, false, false);
            acc6 = __builtin_amdgcn_wmma_i32_16x16x64_iu8(true, A2, true, B1, acc6, false, false);
            acc3 = __builtin_amdgcn_wmma_i32_16x16x64_iu8(true, A3, true, B0, acc3, false, false);
            acc7 = __builtin_amdgcn_wmma_i32_16x16x64_iu8(true, A3, true, B1, acc7, false, false);
        }
    }

    const float scale = (alpha[0] / QMAXF) * (wmax[0] / QMAXF);
    float* ob0 = out + (((size_t)b * 128 + 64) * HD + h) * WD + w0 + n;
    float* ob1 = ob0 + 16;
#pragma unroll
    for (int r = 0; r < 8; ++r) {
        int gb = r + hi * 8;
        float bz0 = bias[gb];
        float bz1 = bias[16 + gb];
        float bz2 = bias[32 + gb];
        float bz3 = bias[48 + gb];
        ob0[(size_t)(gb)*HD * WD]        = fmaxf((float)acc0[r] * scale + bz0, 0.0f);
        ob0[(size_t)(16 + gb) * HD * WD] = fmaxf((float)acc1[r] * scale + bz1, 0.0f);
        ob0[(size_t)(32 + gb) * HD * WD] = fmaxf((float)acc2[r] * scale + bz2, 0.0f);
        ob0[(size_t)(48 + gb) * HD * WD] = fmaxf((float)acc3[r] * scale + bz3, 0.0f);
        ob1[(size_t)(gb)*HD * WD]        = fmaxf((float)acc4[r] * scale + bz0, 0.0f);
        ob1[(size_t)(16 + gb) * HD * WD] = fmaxf((float)acc5[r] * scale + bz1, 0.0f);
        ob1[(size_t)(32 + gb) * HD * WD] = fmaxf((float)acc6[r] * scale + bz2, 0.0f);
        ob1[(size_t)(48 + gb) * HD * WD] = fmaxf((float)acc7[r] * scale + bz3, 0.0f);
    }
}

extern "C" void kernel_launch(void* const* d_in, const int* in_sizes, int n_in,
                              void* d_out, int out_size, void* d_ws, size_t ws_size,
                              hipStream_t stream) {
    const float* x      = (const float*)d_in[0];
    const float* weight = (const float*)d_in[1];
    const float* bias   = (const float*)d_in[2];
    const float* alpha  = (const float*)d_in[3];
    float* out = (float*)d_out;

    float*       ws_wmax = (float*)d_ws;
    signed char* ws_wq   = (signed char*)d_ws + 64;
    signed char* ws_xq   = (signed char*)d_ws + 40960;

    wmax_kernel<<<1, 256, 0, stream>>>(weight, ws_wmax);
    packw_kernel<<<(WQ_BYTES + 255) / 256, 256, 0, stream>>>(weight, ws_wmax, ws_wq);
    zero_kernel<<<2048, 256, 0, stream>>>((int4*)ws_xq, XQ_TOTAL / 16);
    quantx_kernel<<<(NB * HD * WD) / 256, 256, 0, stream>>>(x, alpha, ws_xq, out);
    conv_wmma_kernel<<<1024, 256, 0, stream>>>(ws_xq, ws_wq, bias, alpha, ws_wmax, out);
}
